// MMSingleStreamBlock_61615600828869
// MI455X (gfx1250) — matrix operand order repaired and checked by reading, not compiled, not run on hardware
//
#include <hip/hip_runtime.h>
#include <cstdint>
#include <cstddef>

// ---------------------------------------------------------------- constants
#define HID   3072
#define NH    24
#define HD    128
#define MLP   12288
#define QKV   (3*HID)          // 9216
#define N1    (QKV + MLP)      // 21504
#define CN    (HID + MLP)      // 15360
#define IMG   2048
#define TXT   256
#define SEQ   (IMG + TXT)      // 2304
#define LNEPS 1e-6f

typedef __bf16 bf16;
typedef __attribute__((ext_vector_type(16))) __bf16    v16bf;
typedef __attribute__((ext_vector_type(8)))  __bf16    v8bf;
typedef __attribute__((ext_vector_type(8)))  float     v8f;
typedef __attribute__((ext_vector_type(4)))  uint32_t  u32x4;
typedef __attribute__((ext_vector_type(8)))  uint32_t  u32x8;

// ---------------------------------------------------------------- helpers
__device__ __forceinline__ v8f wmma_bf16(v16bf a, v16bf b, v8f c) {
  // D = A(16x32 bf16) * B(32x16 bf16) + C(16x16 f32)
  return __builtin_amdgcn_wmma_f32_16x16x32_bf16(
      /*neg_a=*/false, a, /*neg_b=*/false, b,
      /*c_mod=*/(short)0, c, /*reuse_a=*/false, /*reuse_b=*/false);
}

// A-matrix fragment (16x32, 16-bit): lane L holds row M=L&15;
// lanes 0-15: K = 0-7 (v0-3) and 16-23 (v4-7); lanes 16-31: K = 8-15, 24-31.
__device__ __forceinline__ v16bf load_frag_a(const bf16* base, int ld) {
  const int lane = threadIdx.x & 31;
  const int m  = lane & 15;
  const int kh = lane >> 4;
  const bf16* p = base + (size_t)m * ld + kh * 8;
  v8bf lo = *(const v8bf*)(p);        // K = kh*8 .. kh*8+7
  v8bf hi = *(const v8bf*)(p + 16);   // K = kh*8+16 .. kh*8+23
  return __builtin_shufflevector(lo, hi, 0,1,2,3,4,5,6,7,8,9,10,11,12,13,14,15);
}

// B-matrix fragment (32x16, 16-bit): lane L holds column N=L&15;
// lanes 0-15: K = 0-7 and 16-23; lanes 16-31: K = 8-15, 24-31.
__device__ __forceinline__ v16bf load_frag_b(const bf16* base, int ld) {
  const int lane = threadIdx.x & 31;
  const int n  = lane & 15;
  const int kh = lane >> 4;
  const bf16* p = base + (size_t)n * ld + kh * 8;
  v8bf lo = *(const v8bf*)(p);
  v8bf hi = *(const v8bf*)(p + 16);
  return __builtin_shufflevector(lo, hi, 0,1,2,3,4,5,6,7,8,9,10,11,12,13,14,15);
}

// CDNA5 async global->LDS copy (tracked by ASYNCcnt), 16B per lane.
#define ASYNC_B128(ldsoff, gaddr)                                             \
  asm volatile("global_load_async_to_lds_b128 %0, %1, off"                    \
               :: "v"(ldsoff), "v"(gaddr) : "memory")
#define ASYNC_WAIT_ALL()  asm volatile("s_wait_asynccnt 0x0" ::: "memory")

__device__ __forceinline__ uint32_t lds_off(const void* p) {
  return (uint32_t)(uintptr_t)p;   // low 32 bits of generic ptr = LDS offset
}

// ------------------------------------------------ Tensor Data Mover (TDM)
// One tensor_load_to_lds DMAs a (tile_dim1=128 rows) x (tile_dim0=32 elems)
// bf16 tile from a row-major 2D tensor into LDS, padding each 64-B row with
// 16 B (pad_interval code 3 = 16 DWORDs, pad_amount code 3 = 4 DWORDs) so the
// LDS row pitch is 40 elements = ROWP. Tracked by TENSORcnt.
__device__ __forceinline__ void tdm_load_2d(const bf16* gaddr, uint32_t ldsaddr,
                                            uint32_t k_len, uint32_t rows_total,
                                            uint32_t stride_elems) {
  const uint64_t ga = (uint64_t)gaddr;
  u32x4 g0;
  g0[0] = 1u;                                        // count=1, user mode
  g0[1] = ldsaddr;                                   // lds_addr
  g0[2] = (uint32_t)ga;                              // global_addr[31:0]
  g0[3] = (uint32_t)(ga >> 32) | 0x80000000u;        // addr[56:32] | type=2
  u32x8 g1;
  g1[0] = (1u << 16) | (1u << 20) | (3u << 22) | (3u << 25); // 2B, pad en
  g1[1] = (k_len & 0xFFFFu) << 16;                   // tensor_dim0 lo16
  g1[2] = (k_len >> 16) | ((rows_total & 0xFFFFu) << 16); // dim0 hi / dim1 lo
  g1[3] = (rows_total >> 16) | (32u << 16);          // dim1 hi | tile_dim0=32
  g1[4] = 128u;                                      // tile_dim1=128, dim2=0
  g1[5] = stride_elems;                              // tensor_dim0_stride lo32
  g1[6] = 0u;                                        // stride hi, dim1_stride
  g1[7] = 0u;
  asm volatile("tensor_load_to_lds %0, %1" :: "s"(g0), "s"(g1) : "memory");
}

__device__ __forceinline__ float gelu_tanh(float x) {
  float x3 = x * x * x;
  return 0.5f * x * (1.f + tanhf(0.7978845608028654f * (x + 0.044715f * x3)));
}

// ---------------------------------------------------------------- fp32 -> bf16
__global__ __launch_bounds__(256)
void f2bf_kernel(const float* __restrict__ in, bf16* __restrict__ out, size_t n) {
  size_t i  = (size_t)blockIdx.x * blockDim.x + threadIdx.x;
  size_t st = (size_t)gridDim.x * blockDim.x;
  for (; i < n; i += st) out[i] = (bf16)in[i];
}

// ---------------------------------------------------------------- mod = silu(vec) @ mod_w.T + mod_b
__global__ __launch_bounds__(256)
void mod_gemv_kernel(const float* __restrict__ vec, const float* __restrict__ mw,
                     const float* __restrict__ mb, float* __restrict__ mod) {
  const int row  = blockIdx.x * 8 + (threadIdx.x >> 5);
  const int lane = threadIdx.x & 31;
  const float* wr = mw + (size_t)row * HID;
  float acc = 0.f;
  for (int c = lane; c < HID; c += 32) {
    float v = vec[c];
    v = v / (1.f + __expf(-v));          // silu
    acc += wr[c] * v;
  }
  #pragma unroll
  for (int off = 16; off; off >>= 1) acc += __shfl_xor(acc, off, 32);
  if (lane == 0) mod[row] = acc + mb[row];
}

// ---------------------------------------------------------------- layernorm + (1+scale)*x + shift -> bf16
__global__ __launch_bounds__(256)
void ln_mod_kernel(const float* __restrict__ x, const float* __restrict__ mod,
                   bf16* __restrict__ xmod) {
  const int tok  = blockIdx.x;
  const int tid  = threadIdx.x;
  const int wave = tid >> 5, lane = tid & 31;
  const float* xr = x + (size_t)tok * HID;
  __shared__ float red[2][8];
  float s = 0.f, s2 = 0.f;
  for (int c = tid; c < HID; c += 256) { float v = xr[c]; s += v; s2 += v * v; }
  #pragma unroll
  for (int off = 16; off; off >>= 1) { s += __shfl_xor(s, off, 32); s2 += __shfl_xor(s2, off, 32); }
  if (lane == 0) { red[0][wave] = s; red[1][wave] = s2; }
  __syncthreads();
  float ts = 0.f, ts2 = 0.f;
  #pragma unroll
  for (int w = 0; w < 8; ++w) { ts += red[0][w]; ts2 += red[1][w]; }
  const float mu  = ts / (float)HID;
  const float var = ts2 / (float)HID - mu * mu;
  const float r   = rsqrtf(var + LNEPS);
  const float* shiftp = mod;
  const float* scalep = mod + HID;
  for (int c = tid; c < HID; c += 256) {
    float v = (xr[c] - mu) * r;
    v = v * (1.f + scalep[c]) + shiftp[c];
    xmod[(size_t)tok * HID + c] = (bf16)v;
  }
}

// ---------------------------------------------------------------- tiled bf16 GEMM: C = A @ B^T (+bias, fused epilogues)
// Staging: Tensor Data Mover (tensor_load_to_lds), double-buffered, wave 0 issues.
#define BM 128
#define BN 128
#define BK 32
#define ROWP 40   // padded LDS row (elements); produced by TDM pad fields

__global__ __launch_bounds__(256)
void gemm_bf16_kernel(const bf16* __restrict__ A, int lda,
                      const bf16* __restrict__ B, int ldb, int K,
                      const float* __restrict__ bias, int mode,
                      bf16* __restrict__ outH,   // mode 0: qkv part  (ld QKV)
                      bf16* __restrict__ outC,   // mode 0: gelu(mlp) -> concat (ld CN)
                      const float* __restrict__ x,
                      const float* __restrict__ gate,
                      float* __restrict__ outF)  // mode 1: residual output (ld HID)
{
  __shared__ __align__(16) bf16 sa[2][BM * ROWP];
  __shared__ __align__(16) bf16 sb[2][BN * ROWP];

  const int tid  = threadIdx.x;
  const int wave = tid >> 5;
  const int lane = tid & 31;
  const int m0 = blockIdx.y * BM;
  const int n0 = blockIdx.x * BN;
  const int wm = (wave & 1) * 64;     // 2 waves in M
  const int wn = (wave >> 1) * 32;    // 4 waves in N
  const uint32_t rowsA = gridDim.y * BM;   // full tensor extents for D#
  const uint32_t rowsB = gridDim.x * BN;

  v8f acc[4][2] = {};

  auto issue = [&](int buf, int kb) {
    tdm_load_2d(A + (size_t)m0 * lda + kb, lds_off(&sa[buf][0]), K, rowsA, lda);
    tdm_load_2d(B + (size_t)n0 * ldb + kb, lds_off(&sb[buf][0]), K, rowsB, ldb);
  };

  if (wave == 0) issue(0, 0);
  const int nk = K / BK;
  for (int kt = 0; kt < nk; ++kt) {
    const int buf = kt & 1;
    if (wave == 0) {
      if (kt + 1 < nk) {
        issue(buf ^ 1, (kt + 1) * BK);
        __builtin_amdgcn_s_wait_tensorcnt((short)2);  // stage kt finished (in-order)
      } else {
        __builtin_amdgcn_s_wait_tensorcnt((short)0);
      }
    }
    __syncthreads();

    v16bf af[4], bfr[2];
    #pragma unroll
    for (int i = 0; i < 4; ++i) af[i]  = load_frag_a(&sa[buf][(wm + i * 16) * ROWP], ROWP);
    #pragma unroll
    for (int j = 0; j < 2; ++j) bfr[j] = load_frag_b(&sb[buf][(wn + j * 16) * ROWP], ROWP);
    #pragma unroll
    for (int i = 0; i < 4; ++i)
      #pragma unroll
      for (int j = 0; j < 2; ++j)
        acc[i][j] = wmma_bf16(af[i], bfr[j], acc[i][j]);
    __syncthreads();   // protect buffer about to be overwritten
  }

  // epilogue: D layout -> VGPR r holds row M = r + 8*(lane>>4), col N = lane&15
  const int half = lane >> 4, coln = lane & 15;
  #pragma unroll
  for (int i = 0; i < 4; ++i)
    #pragma unroll
    for (int j = 0; j < 2; ++j) {
      const int gn = n0 + wn + j * 16 + coln;
      const float bv = bias[gn];
      #pragma unroll
      for (int r = 0; r < 8; ++r) {
        const int gm = m0 + wm + i * 16 + r + 8 * half;
        float v = acc[i][j][r] + bv;
        if (mode == 0) {
          if (gn < QKV) outH[(size_t)gm * QKV + gn] = (bf16)v;
          else          outC[(size_t)gm * CN + (gn - QKV + HID)] = (bf16)gelu_tanh(v);
        } else {
          const size_t o = (size_t)gm * HID + gn;
          outF[o] = x[o] + gate[gn] * v;
        }
      }
    }
}

// ---------------------------------------------------------------- per-head RMSNorm + RoPE (in place on bf16 qkv)
__global__ __launch_bounds__(256)
void qk_prep_kernel(bf16* __restrict__ h, const float* __restrict__ fc,
                    const float* __restrict__ fs, const float* __restrict__ qnw,
                    const float* __restrict__ knw) {
  const int tok  = blockIdx.x;
  const int wave = threadIdx.x >> 5, lane = threadIdx.x & 31;
  const bool img = tok < IMG;
  for (int u = wave; u < 2 * NH; u += 8) {          // (head, q/k) units
    const int head = u >> 1, isk = u & 1;
    bf16* p = h + (size_t)tok * QKV + isk * HID + head * HD;
    const float* w = isk ? knw : qnw;
    float v[4];
    #pragma unroll
    for (int i = 0; i < 4; ++i) v[i] = (float)p[lane * 4 + i];
    float ss = v[0]*v[0] + v[1]*v[1] + v[2]*v[2] + v[3]*v[3];
    #pragma unroll
    for (int off = 16; off; off >>= 1) ss += __shfl_xor(ss, off, 32);
    const float r = rsqrtf(ss / (float)HD + LNEPS);
    float o[4];
    #pragma unroll
    for (int i = 0; i < 4; ++i) o[i] = v[i] * r * w[lane * 4 + i];
    if (img) {
      #pragma unroll
      for (int pi = 0; pi < 2; ++pi) {
        const int d0 = lane * 4 + 2 * pi;
        const float c0 = fc[(size_t)tok * HD + d0], c1 = fc[(size_t)tok * HD + d0 + 1];
        const float s0 = fs[(size_t)tok * HD + d0], s1 = fs[(size_t)tok * HD + d0 + 1];
        const float t1 = o[2 * pi], t2 = o[2 * pi + 1];
        o[2 * pi]     = t1 * c0 - t2 * s0;
        o[2 * pi + 1] = t2 * c1 + t1 * s1;
      }
    }
    #pragma unroll
    for (int i = 0; i < 4; ++i) p[lane * 4 + i] = (bf16)o[i];
  }
}

// ---------------------------------------------------------------- flash attention (WMMA, bf16), 4 waves / block
#define BQ 64
#define BKV 64
#define KPAD (HD + 8)     // 136
#define VPAD (BKV + 8)    // 72

__global__ __launch_bounds__(128)
void attn_kernel(const bf16* __restrict__ h, bf16* __restrict__ concat) {
  const int head = blockIdx.y;
  const int qb   = blockIdx.x * BQ;
  const int tid  = threadIdx.x;
  const int wave = tid >> 5, lane = tid & 31;
  const int half = lane >> 4, coln = lane & 15;
  const float sm_scale = 0.08838834764831845f;   // 1/sqrt(128)

  __shared__ __align__(16) bf16 kt[BKV][KPAD];
  __shared__ __align__(16) bf16 vt[HD][VPAD];          // V transposed: [d][kv]
  __shared__ __align__(16) bf16 pt[4][16][VPAD];       // per-wave P tile

  // Q fragments for this wave's 16 rows (A layout), d = 4 chunks of 32
  v16bf qf[4];
  #pragma unroll
  for (int kc = 0; kc < 4; ++kc)
    qf[kc] = load_frag_a(h + (size_t)(qb + wave * 16) * QKV + head * HD + kc * 32, QKV);

  v8f o[8] = {};
  float mst[8], lst[8];
  #pragma unroll
  for (int r = 0; r < 8; ++r) { mst[r] = -1e30f; lst[r] = 0.f; }

  for (int jt = 0; jt < SEQ / BKV; ++jt) {
    const int kb = jt * BKV;
    __syncthreads();   // previous tile fully consumed before restage

    // K tile -> LDS via async copies (1024 x 16B pieces / 128 threads)
    #pragma unroll
    for (int j = 0; j < 8; ++j) {
      const int idx = tid + 128 * j;
      const int row = idx >> 4, pc = idx & 15;
      uint64_t ga = (uint64_t)(h + (size_t)(kb + row) * QKV + HID + head * HD + pc * 8);
      ASYNC_B128(lds_off(&kt[row][pc * 8]), ga);
    }
    // V tile -> LDS transposed (kv pairs -> b32 stores)
    {
      const int r0 = (tid >> 2) * 2;          // kv pair
      const int db = (tid & 3) * 32;          // d quarter
      const bf16* v0 = h + (size_t)(kb + r0)     * QKV + 2 * HID + head * HD + db;
      const bf16* v1 = h + (size_t)(kb + r0 + 1) * QKV + 2 * HID + head * HD + db;
      #pragma unroll 8
      for (int d = 0; d < 32; ++d) {
        bf16 a = v0[d], b = v1[d];
        uint32_t pk = (uint32_t)__builtin_bit_cast(uint16_t, a) |
                      ((uint32_t)__builtin_bit_cast(uint16_t, b) << 16);
        *(uint32_t*)&vt[db + d][r0] = pk;
      }
    }
    ASYNC_WAIT_ALL();
    __syncthreads();

    // S = Q K^T   (4 tiles of 16 kv columns)
    v8f s[4] = {};
    #pragma unroll
    for (int nt = 0; nt < 4; ++nt)
      #pragma unroll
      for (int kc = 0; kc < 4; ++kc)
        s[nt] = wmma_bf16(qf[kc], load_frag_b(&kt[nt * 16][kc * 32], KPAD), s[nt]);

    // online softmax (row stats per (r, half); reduce across 16 lanes)
    float rmax[8];
    #pragma unroll
    for (int r = 0; r < 8; ++r) {
      float m = fmaxf(fmaxf(s[0][r], s[1][r]), fmaxf(s[2][r], s[3][r]));
      rmax[r] = m * sm_scale;
    }
    #pragma unroll
    for (int off = 8; off; off >>= 1)
      #pragma unroll
      for (int r = 0; r < 8; ++r) rmax[r] = fmaxf(rmax[r], __shfl_xor(rmax[r], off, 32));

    float alpha[8], rsum[8];
    #pragma unroll
    for (int r = 0; r < 8; ++r) {
      const float mn = fmaxf(mst[r], rmax[r]);
      alpha[r] = __expf(mst[r] - mn);
      mst[r] = mn; rsum[r] = 0.f;
    }
    #pragma unroll
    for (int nt = 0; nt < 4; ++nt)
      #pragma unroll
      for (int r = 0; r < 8; ++r) {
        const float p = __expf(s[nt][r] * sm_scale - mst[r]);
        rsum[r] += p;
        pt[wave][r + 8 * half][nt * 16 + coln] = (bf16)p;
      }
    #pragma unroll
    for (int off = 8; off; off >>= 1)
      #pragma unroll
      for (int r = 0; r < 8; ++r) rsum[r] += __shfl_xor(rsum[r], off, 32);
    #pragma unroll
    for (int r = 0; r < 8; ++r) lst[r] = lst[r] * alpha[r] + rsum[r];
    #pragma unroll
    for (int dt = 0; dt < 8; ++dt)
      #pragma unroll
      for (int r = 0; r < 8; ++r) o[dt][r] *= alpha[r];

    // O += P V   (wave-local P through LDS to get A layout)
    v16bf pf0 = load_frag_a(&pt[wave][0][0],  VPAD);
    v16bf pf1 = load_frag_a(&pt[wave][0][32], VPAD);
    #pragma unroll
    for (int dt = 0; dt < 8; ++dt) {
      o[dt] = wmma_bf16(pf0, load_frag_b(&vt[dt * 16][0],  VPAD), o[dt]);
      o[dt] = wmma_bf16(pf1, load_frag_b(&vt[dt * 16][32], VPAD), o[dt]);
    }
  }

  float inv[8];
  #pragma unroll
  for (int r = 0; r < 8; ++r) inv[r] = 1.f / lst[r];
  #pragma unroll
  for (int dt = 0; dt < 8; ++dt)
    #pragma unroll
    for (int r = 0; r < 8; ++r) {
      const int tokr = qb + wave * 16 + r + 8 * half;
      concat[(size_t)tokr * CN + head * HD + dt * 16 + coln] = (bf16)(o[dt][r] * inv[r]);
    }
}

// ---------------------------------------------------------------- host launcher
extern "C" void kernel_launch(void* const* d_in, const int* in_sizes, int n_in,
                              void* d_out, int out_size, void* d_ws, size_t ws_size,
                              hipStream_t stream) {
  const float* x   = (const float*)d_in[0];
  const float* vec = (const float*)d_in[1];
  const float* fc  = (const float*)d_in[2];
  const float* fs  = (const float*)d_in[3];
  const float* mw  = (const float*)d_in[4];
  const float* mb  = (const float*)d_in[5];
  const float* w1  = (const float*)d_in[6];
  const float* b1  = (const float*)d_in[7];
  const float* w2  = (const float*)d_in[8];
  const float* b2  = (const float*)d_in[9];
  const float* qnw = (const float*)d_in[10];
  const float* knw = (const float*)d_in[11];
  float* out = (float*)d_out;

  // workspace carve (~354 MB): mod | w1(bf16) | w2(bf16) | xmod | h(qkv bf16) | concat
  char* w = (char*)d_ws;
  auto carve = [&](size_t bytes) -> void* {
    void* p = (void*)w; w += (bytes + 255) & ~(size_t)255; return p;
  };
  float* mod  = (float*)carve((size_t)QKV * 4);
  bf16* w1b   = (bf16*) carve((size_t)N1 * HID * 2);
  bf16* w2b   = (bf16*) carve((size_t)HID * CN * 2);
  bf16* xmod  = (bf16*) carve((size_t)SEQ * HID * 2);
  bf16* hq    = (bf16*) carve((size_t)SEQ * QKV * 2);
  bf16* cc    = (bf16*) carve((size_t)SEQ * CN * 2);

  // 1) weights -> bf16 (L2-resident GEMM operands)
  f2bf_kernel<<<4096, 256, 0, stream>>>(w1, w1b, (size_t)N1 * HID);
  f2bf_kernel<<<4096, 256, 0, stream>>>(w2, w2b, (size_t)HID * CN);
  // 2) modulation GEMV
  mod_gemv_kernel<<<QKV / 8, 256, 0, stream>>>(vec, mw, mb, mod);
  // 3) layernorm + modulation -> bf16 A
  ln_mod_kernel<<<SEQ, 256, 0, stream>>>(x, mod, xmod);
  // 4) fused QKV+MLP GEMM (gelu fused on MLP half), TDM-staged
  gemm_bf16_kernel<<<dim3(N1 / BN, SEQ / BM), 256, 0, stream>>>(
      xmod, HID, w1b, HID, HID, b1, 0, hq, cc, nullptr, nullptr, nullptr);
  // 5) q/k RMSNorm + RoPE in place
  qk_prep_kernel<<<SEQ, 256, 0, stream>>>(hq, fc, fs, qnw, knw);
  // 6) flash attention -> concat[:, :3072]
  attn_kernel<<<dim3(SEQ / BQ, NH), 128, 0, stream>>>(hq, cc);
  // 7) output GEMM + gate*out residual, TDM-staged
  gemm_bf16_kernel<<<dim3(HID / BN, SEQ / BM), 256, 0, stream>>>(
      cc, CN, w2b, CN, CN, b2, 1, nullptr, nullptr, x, mod + 2 * HID, out);
}
